// LocalAttentionMIL_20650202759874
// MI455X (gfx1250) — matrix operand order repaired and checked by reading, not compiled, not run on hardware
//
#include <hip/hip_runtime.h>
#include <stdint.h>

// ---------------------------------------------------------------------------
// LocalAttentionMIL on MI455X (gfx1250, wave32, WMMA).
// B=4, N=4096, K=16, F=768, D=256, H=8, DH=32, FF=1024, DEPTH=2, M=B*N=16384.
// Dense GEMMs: v_wmma_f32_16x16x32_bf16 fed from LDS, which is filled with
// double-buffered CDNA5 async copies (global_load_async_to_lds_b128 +
// s_wait_asynccnt). Block tile 128Mx64N, wave tile 32Mx32N (4 accumulators).
// ---------------------------------------------------------------------------

#define BATCH 4
#define NTOK  4096
#define MTOK  (BATCH * NTOK)      // 16384 tokens
#define FDIM  768
#define DDIM  256
#define HEADS 8
#define DH    32
#define FFDIM 1024
#define KNBR  16

typedef __attribute__((ext_vector_type(16))) __bf16        v16bf;
typedef __attribute__((ext_vector_type(8)))  float         v8f;
typedef __attribute__((ext_vector_type(4)))  unsigned int  u32x4;

union AFrag { v16bf v; u32x4 u[2]; };

// CDNA5 async global->LDS copy, 16 bytes per lane. Tracked by ASYNCcnt.
__device__ __forceinline__ void async_ld16(unsigned lds_off, const void* gptr) {
  asm volatile("global_load_async_to_lds_b128 %0, %1, off"
               :: "v"(lds_off), "v"(gptr) : "memory");
}

// ---------------------------------------------------------------------------
// Pack a row-major f32 weight W[Kd][Nd] into bf16 WMMA B-fragments.
// Layout per 16(N)x32(K) tile: [tile][lane(32)][elem(16)] contiguous, where
//   lane = 16*kHalf + n (n=0..15), elem e -> K = 16*kHalf + e  (contiguous k order, as the A reads)
// tile index = nTile * (Kd/32) + kStep.
// ---------------------------------------------------------------------------
__global__ __launch_bounds__(256) void pack_b_kernel(
    const float* __restrict__ W, __bf16* __restrict__ P, int Kd, int Nd) {
  int idx = blockIdx.x * 256 + threadIdx.x;
  if (idx >= Kd * Nd) return;
  int e     = idx & 15;
  int lane  = (idx >> 4) & 31;
  int tile  = idx >> 9;
  int ksteps = Kd >> 5;
  int nt = tile / ksteps;
  int ks = tile - nt * ksteps;
  int n    = nt * 16 + (lane & 15);
  int half = lane >> 4;
  int k    = ks * 32 + half * 16 + e;
  P[idx] = (__bf16)W[(size_t)k * Nd + n];
}

// ---------------------------------------------------------------------------
// LayerNorm over F=768, emit bf16. One 256-thread block per row.
// ---------------------------------------------------------------------------
__global__ __launch_bounds__(256) void ln_bf16_kernel(
    const float* __restrict__ X, const float* __restrict__ g,
    const float* __restrict__ bb, __bf16* __restrict__ Y) {
  int row = blockIdx.x;
  int tid = threadIdx.x;
  const float* x = X + (size_t)row * FDIM;
  float v0 = x[tid], v1 = x[tid + 256], v2 = x[tid + 512];
  __shared__ float rs[256], rq[256];
  rs[tid] = v0 + v1 + v2;
  rq[tid] = v0 * v0 + v1 * v1 + v2 * v2;
  __syncthreads();
  for (int off = 128; off > 0; off >>= 1) {
    if (tid < off) { rs[tid] += rs[tid + off]; rq[tid] += rq[tid + off]; }
    __syncthreads();
  }
  float mean = rs[0] * (1.0f / FDIM);
  float var  = rq[0] * (1.0f / FDIM) - mean * mean;
  float inv  = rsqrtf(var + 1e-5f);
  __bf16* y = Y + (size_t)row * FDIM;
  y[tid]       = (__bf16)((v0 - mean) * inv * g[tid]       + bb[tid]);
  y[tid + 256] = (__bf16)((v1 - mean) * inv * g[tid + 256] + bb[tid + 256]);
  y[tid + 512] = (__bf16)((v2 - mean) * inv * g[tid + 512] + bb[tid + 512]);
}

// ---------------------------------------------------------------------------
// WMMA GEMM with async-LDS staging.
// Block = 8 waves -> 128(M) x 64(N); wave grid 4(M) x 2(N); wave tile 32x32
// (4 accumulators, 2 A frags x 2 B frags -> 4 WMMAs per K-step).
// Per K-step: A tile 128x32 bf16 (8KB) + 4 packed B fragment tiles (4KB) are
// async-copied into double-buffered LDS; each of the 256 threads issues
// exactly 3 b128 async loads per stage (pipelined wait: s_wait_asynccnt 3).
// Epilogue EPI:
//   0: outF = acc + bias                               (QKV; outF pre-offset)
//   1: outF = acc + bias + residual                    (Wo + residual x)
//   2: outB = bf16(mask ? 0 : relu(acc + bias))        (FFN1)
//   3: outF = (mask ? 0 : acc + bias) + residual       (FFN2 + residual x1)
// ---------------------------------------------------------------------------
template <int EPI>
__global__ __launch_bounds__(256) void gemm_bf16_wmma(
    const __bf16* __restrict__ A, const __bf16* __restrict__ Bp,
    const float* __restrict__ bias, const float* __restrict__ residual,
    const unsigned char* __restrict__ maskp,
    float* __restrict__ outF, __bf16* __restrict__ outB,
    int Kd, int Nd, int outStride, int Astride) {
  __shared__ __bf16 sA[2][128 * 32];  // row-major: r*32 + k   (8KB per buf)
  __shared__ __bf16 sB[2][4 * 512];   // 4 fragment tiles       (4KB per buf)

  int t    = threadIdx.x;
  int w    = t >> 5;                  // wave 0..7
  int lane = t & 31;
  int half = lane >> 4;
  int wm   = w >> 1;                  // 0..3 (M)
  int wn   = w & 1;                   // 0..1 (N)
  int nBlk = Nd >> 6;                 // N / 64
  int mb   = blockIdx.x / nBlk;
  int nbI  = blockIdx.x - mb * nBlk;
  int ksteps = Kd >> 5;

  // Per-thread async-stage addressing (A rows r and r+64, one B chunk).
  int r = t >> 2, ck = t & 3;
  const __bf16* gA0 = A + (size_t)(mb * 128 + r) * Astride + ck * 8;
  const __bf16* gA1 = gA0 + (size_t)64 * Astride;
  const __bf16* gB0 = Bp + ((size_t)(nbI * 4 + (t >> 6)) * ksteps) * 512 + (t & 63) * 8;
  unsigned lA0 = (unsigned)(uintptr_t)(&sA[0][0] + r * 32 + ck * 8);
  unsigned lA1 = lA0 + 64 * 32 * 2;                    // bytes
  unsigned lB  = (unsigned)(uintptr_t)(&sB[0][0] + (t >> 6) * 512 + (t & 63) * 8);
  const unsigned strA = 128 * 32 * 2;                  // bytes between A bufs
  const unsigned strB = 4 * 512 * 2;                   // bytes between B bufs

  auto issue = [&](int ks, int buf) {
    async_ld16(lA0 + buf * strA, gA0 + (size_t)ks * 32);
    async_ld16(lA1 + buf * strA, gA1 + (size_t)ks * 32);
    async_ld16(lB + buf * strB, gB0 + (size_t)ks * 512);
  };

  // Fragment read addressing (from LDS).
  const __bf16* aRow0 = &sA[0][(wm * 32 + (lane & 15)) * 32];       // M-tile 0
  const __bf16* aRow1 = aRow0 + 16 * 32;                             // M-tile 1
  const __bf16* bT0   = &sB[0][(wn * 2 + 0) * 512 + lane * 16];
  const __bf16* bT1   = &sB[0][(wn * 2 + 1) * 512 + lane * 16];

  v8f c00 = {}; v8f c01 = {}; v8f c10 = {}; v8f c11 = {};
  auto compute = [&](int buf) {
    AFrag a0, a1, b0, b1;
    const __bf16* p;
    p = aRow0 + (size_t)buf * 128 * 32;
    a0.u[0] = *(const u32x4*)(p + half * 16);
    a0.u[1] = *(const u32x4*)(p + 8 + half * 16);
    p = aRow1 + (size_t)buf * 128 * 32;
    a1.u[0] = *(const u32x4*)(p + half * 16);
    a1.u[1] = *(const u32x4*)(p + 8 + half * 16);
    const u32x4* q0 = (const u32x4*)(bT0 + (size_t)buf * 2048);
    b0.u[0] = q0[0]; b0.u[1] = q0[1];
    const u32x4* q1 = (const u32x4*)(bT1 + (size_t)buf * 2048);
    b1.u[0] = q1[0]; b1.u[1] = q1[1];
    c00 = __builtin_amdgcn_wmma_f32_16x16x32_bf16(false, a0.v, false, b0.v,
                                                  (short)0, c00, false, false);
    c01 = __builtin_amdgcn_wmma_f32_16x16x32_bf16(false, a0.v, false, b1.v,
                                                  (short)0, c01, false, false);
    c10 = __builtin_amdgcn_wmma_f32_16x16x32_bf16(false, a1.v, false, b0.v,
                                                  (short)0, c10, false, false);
    c11 = __builtin_amdgcn_wmma_f32_16x16x32_bf16(false, a1.v, false, b1.v,
                                                  (short)0, c11, false, false);
  };

  issue(0, 0);
  for (int ks = 0; ks < ksteps - 1; ++ks) {           // steady state (ksteps>=2)
    int cur = ks & 1;
    issue(ks + 1, cur ^ 1);
    asm volatile("s_wait_asynccnt 0x3" ::: "memory");  // prev stage landed
    __syncthreads();
    compute(cur);
    __syncthreads();
  }
  asm volatile("s_wait_asynccnt 0x0" ::: "memory");
  __syncthreads();
  compute((ksteps - 1) & 1);

  // C layout: VGPR v, lane L -> row = 16*mt + v + 8*(L/16), col = base + L%16
  int mt0 = mb * 8 + wm * 2;            // two stacked 16-row M-tiles
  int np  = nbI * 2 + wn;
  int colBase = np * 32 + (lane & 15);
  for (int pair = 0; pair < 2; ++pair) {
    v8f cc0 = pair ? c10 : c00;
    v8f cc1 = pair ? c11 : c01;
    int mt = mt0 + pair;
    for (int v = 0; v < 8; ++v) {
      int rr = mt * 16 + v + (half << 3);
      int cA = colBase, cB = colBase + 16;
      float a0 = cc0[v] + bias[cA];
      float a1 = cc1[v] + bias[cB];
      if (EPI == 0) {
        outF[(size_t)rr * outStride + cA] = a0;
        outF[(size_t)rr * outStride + cB] = a1;
      } else if (EPI == 1) {
        size_t o0 = (size_t)rr * outStride + cA, o1 = (size_t)rr * outStride + cB;
        outF[o0] = a0 + residual[o0];
        outF[o1] = a1 + residual[o1];
      } else if (EPI == 2) {
        float mk = maskp[rr] ? 0.0f : 1.0f;
        outB[(size_t)rr * outStride + cA] = (__bf16)(fmaxf(a0, 0.0f) * mk);
        outB[(size_t)rr * outStride + cB] = (__bf16)(fmaxf(a1, 0.0f) * mk);
      } else {
        float mk = maskp[rr] ? 0.0f : 1.0f;
        size_t o0 = (size_t)rr * outStride + cA, o1 = (size_t)rr * outStride + cB;
        outF[o0] = a0 * mk + residual[o0];
        outF[o1] = a1 * mk + residual[o1];
      }
    }
  }
}

// ---------------------------------------------------------------------------
// Local attention: one wave per token. qkv[M][768] = [q|k|v] f32.
// Emits bf16 attention output [M][256] for the Wo WMMA GEMM.
// ---------------------------------------------------------------------------
__global__ __launch_bounds__(256) void attn_kernel(
    const float* __restrict__ qkv, const int* __restrict__ nbr,
    const unsigned char* __restrict__ maskp, __bf16* __restrict__ outB) {
  int slot = threadIdx.x >> 5;
  int lane = threadIdx.x & 31;
  int m = (blockIdx.x << 3) + slot;
  int b = m >> 12;  // / NTOK

  __shared__ int   s_nb[8][KNBR];
  __shared__ float s_attn[8][HEADS][KNBR];
  if (lane < KNBR) s_nb[slot][lane] = nbr[(size_t)m * KNBR + lane];
  __syncthreads();

  const float* q = qkv + (size_t)m * FDIM;
  const float scale = 0.17677669529663687f;  // 1/sqrt(32)

  // 128 scores (h,k) -> 4 per lane: s = 32*i + lane, h = s/16, k = s%16
  float sc[4];
  for (int i = 0; i < 4; ++i) {
    int s = (i << 5) + lane;
    int h = s >> 4, k = s & 15;
    int nb = s_nb[slot][k];
    const float4* kr = (const float4*)(qkv + ((size_t)(b * NTOK + nb)) * FDIM +
                                       DDIM + h * DH);
    const float4* qh = (const float4*)(q + h * DH);
    float acc = 0.0f;
    for (int d = 0; d < DH / 4; ++d) {
      float4 a = qh[d], bb = kr[d];
      acc += a.x * bb.x + a.y * bb.y + a.z * bb.z + a.w * bb.w;
    }
    acc *= scale;
    if (maskp[b * NTOK + nb]) acc = -1e9f;
    sc[i] = acc;
  }
  // softmax over k: 16-lane groups (fixed h lives in one half-wave per i)
  for (int i = 0; i < 4; ++i) {
    float mx = sc[i];
    for (int off = 8; off > 0; off >>= 1) mx = fmaxf(mx, __shfl_xor(mx, off, 16));
    float e = __expf(sc[i] - mx);
    float sum = e;
    for (int off = 8; off > 0; off >>= 1) sum += __shfl_xor(sum, off, 16);
    int s = (i << 5) + lane;
    s_attn[slot][s >> 4][s & 15] = e / sum;
  }
  __syncthreads();

  // out[c] = sum_k attn[h][k] * v_nb[k][c];  c = 32*j + lane -> h=j, d=lane
  for (int j = 0; j < HEADS; ++j) {
    int c = (j << 5) + lane;
    float acc = 0.0f;
    for (int k = 0; k < KNBR; ++k) {
      int nb = s_nb[slot][k];
      acc += s_attn[slot][j][k] *
             qkv[((size_t)(b * NTOK + nb)) * FDIM + 2 * DDIM + c];
    }
    outB[(size_t)m * DDIM + c] = (__bf16)acc;
  }
}

// ---------------------------------------------------------------------------
// Final head: logits = mask ? 0 : x @ Wm + bm.  One wave per token.
// ---------------------------------------------------------------------------
__global__ __launch_bounds__(256) void head_kernel(
    const float* __restrict__ X, const float* __restrict__ Wm,
    const float* __restrict__ bm, const unsigned char* __restrict__ maskp,
    float* __restrict__ out) {
  int lane = threadIdx.x & 31;
  int m = (blockIdx.x << 3) + (threadIdx.x >> 5);
  const float* x = X + (size_t)m * FDIM;
  float acc = 0.0f;
  for (int t = lane; t < FDIM; t += 32) acc += x[t] * Wm[t];
  for (int off = 16; off > 0; off >>= 1) acc += __shfl_xor(acc, off, 32);
  if (lane == 0) out[m] = maskp[m] ? 0.0f : (acc + bm[0]);
}

// ---------------------------------------------------------------------------
// Host orchestration.
// Input order (top-level dict insertion order; params leaves sorted-key):
//  0:x  1:neighbors  2:mask  3:Wm  4:bm
//  5+16*l+j, j: 0 W1, 1 W2, 2 Wk, 3 Wo, 4 Wq, 5 Wv, 6 b1, 7 b2, 8 bk, 9 bo,
//              10 bq, 11 bv, 12 ln1_b, 13 ln1_g, 14 ln2_b, 15 ln2_g
// ---------------------------------------------------------------------------
extern "C" void kernel_launch(void* const* d_in, const int* in_sizes, int n_in,
                              void* d_out, int out_size, void* d_ws,
                              size_t ws_size, hipStream_t stream) {
  (void)in_sizes; (void)n_in; (void)out_size; (void)ws_size;
  const float*         x_in  = (const float*)d_in[0];
  const int*           nbr   = (const int*)d_in[1];
  const unsigned char* maskp = (const unsigned char*)d_in[2];
  const float*         Wm    = (const float*)d_in[3];
  const float*         bm    = (const float*)d_in[4];
  auto L = [&](int l, int j) { return (const float*)d_in[5 + 16 * l + j]; };

  // Workspace bump allocator (256B aligned).
  char* ws = (char*)d_ws;
  size_t off = 0;
  auto alloc = [&](size_t bytes) {
    off = (off + 255) & ~(size_t)255;
    void* p = ws + off;
    off += bytes;
    return p;
  };
  __bf16* pQ[2]; __bf16* pK[2]; __bf16* pV[2];
  __bf16* pO[2]; __bf16* p1[2]; __bf16* p2[2];
  for (int l = 0; l < 2; ++l) {
    pQ[l] = (__bf16*)alloc((size_t)FDIM * DDIM * 2);
    pK[l] = (__bf16*)alloc((size_t)FDIM * DDIM * 2);
    pV[l] = (__bf16*)alloc((size_t)FDIM * DDIM * 2);
    pO[l] = (__bf16*)alloc((size_t)DDIM * FDIM * 2);
    p1[l] = (__bf16*)alloc((size_t)FDIM * FFDIM * 2);
    p2[l] = (__bf16*)alloc((size_t)FFDIM * FDIM * 2);
  }
  float*  x_ws  = (float*)alloc((size_t)MTOK * FDIM * 4);   // residual stream
  float*  x1_ws = (float*)alloc((size_t)MTOK * FDIM * 4);   // post-attn stream
  float*  qkv   = (float*)alloc((size_t)MTOK * FDIM * 4);   // [q|k|v]
  __bf16* xb    = (__bf16*)alloc((size_t)MTOK * FDIM * 2);  // LN output
  __bf16* attnb = (__bf16*)alloc((size_t)MTOK * DDIM * 2);
  __bf16* hb    = (__bf16*)alloc((size_t)MTOK * FFDIM * 2);

  // Pack all weights to bf16 WMMA fragments.
  for (int l = 0; l < 2; ++l) {
    pack_b_kernel<<<(FDIM * DDIM) / 256, 256, 0, stream>>>(L(l, 4), pQ[l], FDIM, DDIM);
    pack_b_kernel<<<(FDIM * DDIM) / 256, 256, 0, stream>>>(L(l, 2), pK[l], FDIM, DDIM);
    pack_b_kernel<<<(FDIM * DDIM) / 256, 256, 0, stream>>>(L(l, 5), pV[l], FDIM, DDIM);
    pack_b_kernel<<<(DDIM * FDIM) / 256, 256, 0, stream>>>(L(l, 3), pO[l], DDIM, FDIM);
    pack_b_kernel<<<(FDIM * FFDIM) / 256, 256, 0, stream>>>(L(l, 0), p1[l], FDIM, FFDIM);
    pack_b_kernel<<<(FFDIM * FDIM) / 256, 256, 0, stream>>>(L(l, 1), p2[l], FFDIM, FDIM);
  }

  const int MB = MTOK / 128;  // 128 M-blocks
  for (int l = 0; l < 2; ++l) {
    const float* bq = L(l, 10); const float* bk = L(l, 8);
    const float* bv = L(l, 11); const float* bo = L(l, 9);
    const float* b1 = L(l, 6);  const float* b2 = L(l, 7);
    const float* xin = (l == 0) ? x_in : x_ws;

    // x1 = LN1(x) -> bf16
    ln_bf16_kernel<<<MTOK, 256, 0, stream>>>(xin, L(l, 13), L(l, 12), xb);
    // QKV GEMMs into qkv[M][768] at column offsets 0/256/512
    gemm_bf16_wmma<0><<<MB * (DDIM / 64), 256, 0, stream>>>(
        xb, pQ[l], bq, nullptr, nullptr, qkv + 0, nullptr, FDIM, DDIM, FDIM, FDIM);
    gemm_bf16_wmma<0><<<MB * (DDIM / 64), 256, 0, stream>>>(
        xb, pK[l], bk, nullptr, nullptr, qkv + DDIM, nullptr, FDIM, DDIM, FDIM, FDIM);
    gemm_bf16_wmma<0><<<MB * (DDIM / 64), 256, 0, stream>>>(
        xb, pV[l], bv, nullptr, nullptr, qkv + 2 * DDIM, nullptr, FDIM, DDIM, FDIM, FDIM);
    // local attention -> bf16 [M][256]
    attn_kernel<<<MTOK / 8, 256, 0, stream>>>(qkv, nbr, maskp, attnb);
    // x1_ws = attn @ Wo + bo + x
    gemm_bf16_wmma<1><<<MB * (FDIM / 64), 256, 0, stream>>>(
        attnb, pO[l], bo, xin, nullptr, x1_ws, nullptr, DDIM, FDIM, FDIM, DDIM);
    // x2 = LN2(x1) -> bf16
    ln_bf16_kernel<<<MTOK, 256, 0, stream>>>(x1_ws, L(l, 15), L(l, 14), xb);
    // h = mask(relu(x2 @ W1 + b1)) -> bf16
    gemm_bf16_wmma<2><<<MB * (FFDIM / 64), 256, 0, stream>>>(
        xb, p1[l], b1, nullptr, maskp, nullptr, hb, FDIM, FFDIM, FFDIM, FDIM);
    // x = mask(h @ W2 + b2) + x1
    gemm_bf16_wmma<3><<<MB * (FDIM / 64), 256, 0, stream>>>(
        hb, p2[l], b2, x1_ws, maskp, x_ws, nullptr, FFDIM, FDIM, FDIM, FFDIM);
  }

  head_kernel<<<MTOK / 8, 256, 0, stream>>>(x_ws, Wm, bm, maskp, (float*)d_out);
}
